// Decoder_48163763257917
// MI455X (gfx1250) — compile-verified, matrix operand order checked by reading, not run on hardware
//
#include <hip/hip_runtime.h>

typedef __attribute__((ext_vector_type(16))) _Float16 v16h;
typedef __attribute__((ext_vector_type(8)))  float    v8f;
typedef _Float16 f16;

#define B_      32
#define T_IN_   512
#define T_OUT_  600
#define ENC_    512
#define HID_    1024     // ATT_RNN == DEC_RNN
#define G4_     4096     // 4*HID
#define ATTD_   128
#define NFILT_  32
#define KSZ_    31
#define PAD_    15
#define KA_     1792     // 256 prenet + 512 ctx + 1024 h
#define KD_     2560     // 1024 ah + 512 ctx + 1024 dh
#define KP_     1536     // 1024 dh + 512 ctx
#define KF_     512

__device__ __forceinline__ float sigmoidf_(float x) { return 1.0f / (1.0f + __expf(-x)); }

// ---------------------------------------------------------------------------
// WMMA GEMM: C[32 x N] = A[32 x K] (f16 row-major) * Bpacked + bias
// B pre-packed in WMMA-native tile order: tile (ntile*nkt + kt) = 512 f16,
// element offset = lane*16 + i  (one contiguous v16h per lane).
// Each wave owns ONE M-tile and TWO N-tiles -> two independent accumulator
// chains to hide the f16 WMMA->WMMA RAW hazard (ISA 7.12.1), and one shared
// A-fragment load (two contiguous uint4 runs per lane, derived from the
// 16-bit A layout in ISA 7.12.2).
// Grid: N/128 blocks, 256 threads (8 waves = 2 mtiles x 4 npairs).
// ---------------------------------------------------------------------------
__global__ __launch_bounds__(256)
void wmma_gemm32(const f16* __restrict__ A, int lda,
                 const f16* __restrict__ Bpk,
                 const float* __restrict__ bias,
                 float* __restrict__ C, long ldc,
                 f16* __restrict__ Ch, int ldch,
                 int N, int K)
{
    const int lane   = threadIdx.x & 31;
    const int wave   = threadIdx.x >> 5;
    const int mtile  = wave & 1;
    const int ntile0 = blockIdx.x * 8 + (wave >> 1) * 2;   // two N-tiles per wave
    const int ntile1 = ntile0 + 1;
    if (ntile0 * 16 >= N) return;

    const int nkt   = K >> 5;           // K tiles of 32
    const int half_ = lane >> 4;
    const int mrow  = mtile * 16 + (lane & 15);

    const uint32_t* Arow = (const uint32_t*)A + (size_t)mrow * (lda >> 1);
    const v16h* Bp = (const v16h*)Bpk;
    size_t btl0 = (size_t)ntile0 * nkt * 32 + lane;
    size_t btl1 = (size_t)ntile1 * nkt * 32 + lane;

    v8f acc0 = {};
    v8f acc1 = {};
    for (int kt = 0; kt < nkt; ++kt) {
        // A fragment: per-lane K indices form two contiguous 4-dword runs
        const uint4* Ak4 = (const uint4*)(Arow + kt * 16);
        uint4 lo = Ak4[half_];
        uint4 hi = Ak4[2 + half_];
        union { v16h v; uint4 q[2]; } a;
        a.q[0] = lo; a.q[1] = hi;

        v16h b0 = Bp[btl0 + (size_t)kt * 32];
        v16h b1 = Bp[btl1 + (size_t)kt * 32];
        if (kt + 1 < nkt) {
            __builtin_prefetch((const void*)&Bp[btl0 + (size_t)(kt + 1) * 32], 0, 1);
            __builtin_prefetch((const void*)&Bp[btl1 + (size_t)(kt + 1) * 32], 0, 1);
        }

        acc0 = __builtin_amdgcn_wmma_f32_16x16x32_f16(
            false, a.v, false, b0, (short)0, acc0, false, false);
        acc1 = __builtin_amdgcn_wmma_f32_16x16x32_f16(
            false, a.v, false, b1, (short)0, acc1, false, false);
    }

    const int nloc  = lane & 15;
    const int ncol0 = ntile0 * 16;
    const int ncol1 = ntile1 * 16;
    const float bt0 = bias ? bias[ncol0 + nloc] : 0.0f;
    const float bt1 = bias ? bias[ncol1 + nloc] : 0.0f;
#pragma unroll
    for (int r = 0; r < 8; ++r) {
        int m = mtile * 16 + r + 8 * half_;          // C layout: lanes16-31 -> M=8+r
        float v0 = acc0[r] + bt0;
        float v1 = acc1[r] + bt1;
        if (C) {
            C[(size_t)m * ldc + ncol0 + nloc] = v0;
            C[(size_t)m * ldc + ncol1 + nloc] = v1;
        }
        if (Ch) {
            Ch[(size_t)m * ldch + ncol0 + nloc] = (f16)v0;
            Ch[(size_t)m * ldch + ncol1 + nloc] = (f16)v1;
        }
    }
}

// ---------------------------------------------------------------------------
// Pack W (row-major f32, rows are output-features N, cols K) into WMMA tile
// order as f16. K is a concat of two sources [W1 (N x K1) | W2 (N x K2)].
// ---------------------------------------------------------------------------
__global__ void pack_weights(const float* __restrict__ W1, int K1,
                             const float* __restrict__ W2, int K2,
                             f16* __restrict__ out, int N, int Ktot)
{
    size_t idx = (size_t)blockIdx.x * blockDim.x + threadIdx.x;
    size_t total = (size_t)N * Ktot;
    if (idx >= total) return;
    int nkt   = Ktot >> 5;
    int i     = (int)(idx & 15);
    int lane  = (int)((idx >> 4) & 31);
    size_t tl = idx >> 9;
    int kt    = (int)(tl % nkt);
    int ntile = (int)(tl / nkt);
    int v = i >> 1, hi = i & 1;
    int half_ = lane >> 4;
    int n = ntile * 16 + (lane & 15);
    int kb = (v < 4) ? (half_ * 8 + 2 * v) : (16 + half_ * 8 + 2 * (v - 4));
    int kk = kt * 32 + kb + hi;
    float val = (kk < K1) ? W1[(size_t)n * K1 + kk]
                          : W2[(size_t)n * K2 + (kk - K1)];
    out[idx] = (f16)val;
}

__global__ void bias_sum(const float* __restrict__ a, const float* __restrict__ b,
                         float* __restrict__ out, int n)
{
    int i = blockIdx.x * blockDim.x + threadIdx.x;
    if (i < n) out[i] = a[i] + b[i];
}

// deterministic hash dropout (p=0.5, scale 2.0)
__device__ __forceinline__ float drop_mask(int s, int b, int j, int stage)
{
    unsigned h = (unsigned)s * 73856093u ^ (unsigned)b * 19349663u ^
                 (unsigned)j * 83492791u ^ (unsigned)(stage + 1) * 2654435761u;
    h ^= h >> 13; h *= 0x5bd1e995u; h ^= h >> 15;
    return (h & 1u) ? 2.0f : 0.0f;
}

// prenet for all 600 steps: out[s][b][j] f16
__global__ __launch_bounds__(256)
void prenet_kernel(const float* __restrict__ emb, const float* __restrict__ w1,
                   const float* __restrict__ w2, const int* __restrict__ units,
                   f16* __restrict__ out)
{
    int s = blockIdx.x, b = blockIdx.y, j = threadIdx.x;
    __shared__ float x0[256], x1[256];
    float xv = 0.0f;
    if (s > 0) {
        int u = units[b * T_OUT_ + (s - 1)];
        xv = emb[(size_t)u * 256 + j];
    }
    x0[j] = xv;
    __syncthreads();
    float acc = 0.0f;
    const float* wr = w1 + (size_t)j * 256;
    for (int k = 0; k < 256; ++k) acc += wr[k] * x0[k];
    x1[j] = fmaxf(acc, 0.0f) * drop_mask(s, b, j, 0);
    __syncthreads();
    float acc2 = 0.0f;
    const float* wr2 = w2 + (size_t)j * 256;
    for (int k = 0; k < 256; ++k) acc2 += wr2[k] * x1[k];
    acc2 = fmaxf(acc2, 0.0f) * drop_mask(s, b, j, 1);
    out[((size_t)s * B_ + b) * 256 + j] = (f16)acc2;
}

// pm[b][t][:] = memory[b][t][:] @ wm^T
__global__ __launch_bounds__(128)
void pm_kernel(const float* __restrict__ memory, const float* __restrict__ wm,
               float* __restrict__ pm)
{
    int bt = blockIdx.x;                       // 0 .. B*T_IN-1
    __shared__ float mrow[ENC_];
    const float* src = memory + (size_t)bt * ENC_;
    for (int k = threadIdx.x; k < ENC_; k += 128) mrow[k] = src[k];
    __syncthreads();
    const float* w = wm + (size_t)threadIdx.x * ENC_;
    float acc = 0.0f;
    for (int k = 0; k < ENC_; ++k) acc += w[k] * mrow[k];
    pm[(size_t)bt * ATTD_ + threadIdx.x] = acc;
}

// copy this step's prenet slice into Xcat_a[:, 0:256]
__global__ void step_copy(const f16* __restrict__ pre, f16* __restrict__ xa)
{
    int idx = blockIdx.x * 256 + threadIdx.x;
    if (idx >= B_ * 256) return;
    int b = idx >> 8, j = idx & 255;
    xa[(size_t)b * KA_ + j] = pre[idx];
}

// LSTM elementwise: gates[32 x 4096] -> h (f32 opt) + f16 copies into Xcat rows
__global__ void lstm_pointwise(const float* __restrict__ gates,
                               float* __restrict__ c,
                               float* __restrict__ h_f32,
                               f16* __restrict__ h1, int ld1,
                               f16* __restrict__ h2, int ld2)
{
    int idx = blockIdx.x * blockDim.x + threadIdx.x;
    if (idx >= B_ * HID_) return;
    int b = idx >> 10, j = idx & 1023;
    const float* g = gates + (size_t)b * G4_;
    float gi = g[j], gf = g[HID_ + j], gg = g[2 * HID_ + j], go = g[3 * HID_ + j];
    float cn = sigmoidf_(gf) * c[idx] + sigmoidf_(gi) * tanhf(gg);
    float hn = sigmoidf_(go) * tanhf(cn);
    c[idx] = cn;
    if (h_f32) h_f32[idx] = hn;
    h1[(size_t)b * ld1 + j] = (f16)hn;
    if (h2) h2[(size_t)b * ld2 + j] = (f16)hn;
}

// location-sensitive attention, one block per batch element
__global__ __launch_bounds__(256)
void attention_kernel(const float* __restrict__ ah,       // 32 x 1024
                      const float* __restrict__ wq,       // 128 x 1024
                      const float* __restrict__ vvec,     // 128
                      const float* __restrict__ loc_conv, // 32 x 2 x 31
                      const float* __restrict__ loc_dense,// 128 x 32
                      const float* __restrict__ pm,       // 32 x 512 x 128
                      const float* __restrict__ memory,   // 32 x 512 x 512
                      const int*   __restrict__ memlen,
                      float* __restrict__ aw, float* __restrict__ awc,
                      float* __restrict__ aligns, int t_step,
                      f16* __restrict__ xa, f16* __restrict__ xd, f16* __restrict__ xp)
{
    int b = blockIdx.x, tid = threadIdx.x;
    __shared__ float s_ah[HID_];
    __shared__ float s_pq[ATTD_];
    __shared__ float s_cw[NFILT_ * 2 * KSZ_];
    __shared__ float s_ld[ATTD_ * NFILT_];
    __shared__ float s_aw[T_IN_], s_awc[T_IN_], s_e[T_IN_];
    __shared__ float s_red[256];

    for (int i = tid; i < HID_; i += 256) s_ah[i] = ah[(size_t)b * HID_ + i];
    for (int i = tid; i < NFILT_ * 2 * KSZ_; i += 256) s_cw[i] = loc_conv[i];
    for (int i = tid; i < ATTD_ * NFILT_; i += 256) s_ld[i] = loc_dense[i];
    for (int i = tid; i < T_IN_; i += 256) {
        s_aw[i]  = aw [(size_t)b * T_IN_ + i];
        s_awc[i] = awc[(size_t)b * T_IN_ + i];
    }
    __syncthreads();
    if (tid < ATTD_) {
        const float* w = wq + (size_t)tid * HID_;
        float acc = 0.0f;
        for (int k = 0; k < HID_; ++k) acc += w[k] * s_ah[k];
        s_pq[tid] = acc;
    }
    __syncthreads();

    int L = memlen[b];
    for (int t = tid; t < T_IN_; t += 256) {
        float co[NFILT_];
#pragma unroll
        for (int f = 0; f < NFILT_; ++f) co[f] = 0.0f;
        for (int k = 0; k < KSZ_; ++k) {
            int ti = t + k - PAD_;
            bool ok = (ti >= 0) && (ti < T_IN_);
            float a0 = ok ? s_aw[ti]  : 0.0f;
            float a1 = ok ? s_awc[ti] : 0.0f;
#pragma unroll
            for (int f = 0; f < NFILT_; ++f)
                co[f] += s_cw[f * 62 + k] * a0 + s_cw[f * 62 + KSZ_ + k] * a1;
        }
        const float* pmrow = pm + ((size_t)b * T_IN_ + t) * ATTD_;
        float e = 0.0f;
        for (int d = 0; d < ATTD_; ++d) {
            float acc = s_pq[d] + pmrow[d];
#pragma unroll
            for (int f = 0; f < NFILT_; ++f) acc += s_ld[d * NFILT_ + f] * co[f];
            e += vvec[d] * tanhf(acc);
        }
        s_e[t] = (t >= L) ? -1e30f : e;
    }
    __syncthreads();

    float m = -1e30f;
    for (int t = tid; t < T_IN_; t += 256) m = fmaxf(m, s_e[t]);
    s_red[tid] = m; __syncthreads();
    for (int s = 128; s > 0; s >>= 1) {
        if (tid < s) s_red[tid] = fmaxf(s_red[tid], s_red[tid + s]);
        __syncthreads();
    }
    m = s_red[0]; __syncthreads();
    float sum = 0.0f;
    for (int t = tid; t < T_IN_; t += 256) { float ex = __expf(s_e[t] - m); s_e[t] = ex; sum += ex; }
    s_red[tid] = sum; __syncthreads();
    for (int s = 128; s > 0; s >>= 1) {
        if (tid < s) s_red[tid] += s_red[tid + s];
        __syncthreads();
    }
    float inv = 1.0f / s_red[0];
    __syncthreads();
    for (int t = tid; t < T_IN_; t += 256) {
        float a = s_e[t] * inv;
        s_aw[t] = a;
        aw [(size_t)b * T_IN_ + t] = a;
        awc[(size_t)b * T_IN_ + t] = s_awc[t] + a;
        aligns[((size_t)b * T_OUT_ + t_step) * T_IN_ + t] = a;
    }
    __syncthreads();
    // ctx[d] = sum_t aw[t] * memory[b][t][d]  (coalesced across threads in d)
    for (int d = tid; d < ENC_; d += 256) {
        float acc = 0.0f;
        const float* mb = memory + (size_t)b * T_IN_ * ENC_ + d;
        for (int t = 0; t < T_IN_; ++t) acc += s_aw[t] * mb[(size_t)t * ENC_];
        f16 ch = (f16)acc;
        xa[(size_t)b * KA_ + 256  + d] = ch;
        xd[(size_t)b * KD_ + 1024 + d] = ch;
        xp[(size_t)b * KP_ + 1024 + d] = ch;
    }
}

// ---------------------------------------------------------------------------
extern "C" void kernel_launch(void* const* d_in, const int* in_sizes, int n_in,
                              void* d_out, int out_size, void* d_ws, size_t ws_size,
                              hipStream_t stream)
{
    (void)in_sizes; (void)n_in; (void)out_size;
    // setup_inputs() insertion order
    const float* memory    = (const float*)d_in[0];
    const float* emb       = (const float*)d_in[1];
    const float* pre_w1    = (const float*)d_in[2];
    const float* pre_w2    = (const float*)d_in[3];
    const float* arnn_wih  = (const float*)d_in[4];
    const float* arnn_whh  = (const float*)d_in[5];
    const float* arnn_bih  = (const float*)d_in[6];
    const float* arnn_bhh  = (const float*)d_in[7];
    const float* wq        = (const float*)d_in[8];
    const float* wm        = (const float*)d_in[9];
    const float* vvec      = (const float*)d_in[10];
    const float* loc_conv  = (const float*)d_in[11];
    const float* loc_dense = (const float*)d_in[12];
    const float* drnn_wih  = (const float*)d_in[13];
    const float* drnn_whh  = (const float*)d_in[14];
    const float* drnn_bih  = (const float*)d_in[15];
    const float* drnn_bhh  = (const float*)d_in[16];
    const float* proj_w    = (const float*)d_in[17];
    const float* proj_b    = (const float*)d_in[18];
    const float* final_w   = (const float*)d_in[19];
    const float* final_b   = (const float*)d_in[20];
    const int*   units     = (const int*)d_in[21];
    const int*   memlen    = (const int*)d_in[22];

    float* logits = (float*)d_out;                                  // (B,T_OUT,512)
    float* aligns = logits + (size_t)B_ * T_OUT_ * 512;             // (B,T_OUT,512)

    // workspace carve-up
    char* ws = (char*)d_ws;
    size_t off = 0;
    auto carve = [&](size_t bytes) { char* p = ws + off; off += (bytes + 255) & ~(size_t)255; return (void*)p; };
    f16*   pkA    = (f16*)  carve((size_t)G4_ * KA_ * 2);
    f16*   pkD    = (f16*)  carve((size_t)G4_ * KD_ * 2);
    f16*   pkP    = (f16*)  carve((size_t)ENC_ * KP_ * 2);
    f16*   pkF    = (f16*)  carve((size_t)512 * KF_ * 2);
    f16*   prenet = (f16*)  carve((size_t)T_OUT_ * B_ * 256 * 2);
    float* pm     = (float*)carve((size_t)B_ * T_IN_ * ATTD_ * 4);
    float* bias_a = (float*)carve((size_t)G4_ * 4);
    float* bias_d = (float*)carve((size_t)G4_ * 4);
    float* gates  = (float*)carve((size_t)B_ * G4_ * 4);
    f16*   xa     = (f16*)  carve((size_t)B_ * KA_ * 2);
    f16*   xd     = (f16*)  carve((size_t)B_ * KD_ * 2);
    f16*   xp     = (f16*)  carve((size_t)B_ * KP_ * 2);
    float* ah     = (float*)carve((size_t)B_ * HID_ * 4);
    float* ac     = (float*)carve((size_t)B_ * HID_ * 4);
    float* dc     = (float*)carve((size_t)B_ * HID_ * 4);
    float* awb    = (float*)carve((size_t)B_ * T_IN_ * 4);
    float* awcb   = (float*)carve((size_t)B_ * T_IN_ * 4);
    f16*   hid    = (f16*)  carve((size_t)B_ * 512 * 2);
    if (off > ws_size) return;

    // zero recurrent state (fresh every call -> deterministic)
    hipMemsetAsync(ah,   0, (size_t)B_ * HID_ * 4, stream);
    hipMemsetAsync(ac,   0, (size_t)B_ * HID_ * 4, stream);
    hipMemsetAsync(dc,   0, (size_t)B_ * HID_ * 4, stream);
    hipMemsetAsync(awb,  0, (size_t)B_ * T_IN_ * 4, stream);
    hipMemsetAsync(awcb, 0, (size_t)B_ * T_IN_ * 4, stream);
    hipMemsetAsync(xa,   0, (size_t)B_ * KA_ * 2, stream);
    hipMemsetAsync(xd,   0, (size_t)B_ * KD_ * 2, stream);
    hipMemsetAsync(xp,   0, (size_t)B_ * KP_ * 2, stream);

    // preprocessing
    {
        size_t tA = (size_t)G4_ * KA_;
        pack_weights<<<(unsigned)((tA + 255) / 256), 256, 0, stream>>>(arnn_wih, 768, arnn_whh, 1024, pkA, G4_, KA_);
        size_t tD = (size_t)G4_ * KD_;
        pack_weights<<<(unsigned)((tD + 255) / 256), 256, 0, stream>>>(drnn_wih, 1536, drnn_whh, 1024, pkD, G4_, KD_);
        size_t tP = (size_t)ENC_ * KP_;
        pack_weights<<<(unsigned)((tP + 255) / 256), 256, 0, stream>>>(proj_w, KP_, nullptr, 1, pkP, ENC_, KP_);
        size_t tF = (size_t)512 * KF_;
        pack_weights<<<(unsigned)((tF + 255) / 256), 256, 0, stream>>>(final_w, KF_, nullptr, 1, pkF, 512, KF_);
        bias_sum<<<G4_ / 256, 256, 0, stream>>>(arnn_bih, arnn_bhh, bias_a, G4_);
        bias_sum<<<G4_ / 256, 256, 0, stream>>>(drnn_bih, drnn_bhh, bias_d, G4_);
        prenet_kernel<<<dim3(T_OUT_, B_), 256, 0, stream>>>(emb, pre_w1, pre_w2, units, prenet);
        pm_kernel<<<B_ * T_IN_, 128, 0, stream>>>(memory, wm, pm);
    }

    // decode loop (replayed via graph capture)
    for (int t = 0; t < T_OUT_; ++t) {
        step_copy<<<(B_ * 256) / 256, 256, 0, stream>>>(prenet + (size_t)t * B_ * 256, xa);

        // attention LSTM gates: [prenet|ctx|ah] @ [wih|whh]^T
        wmma_gemm32<<<G4_ / 128, 256, 0, stream>>>(xa, KA_, pkA, bias_a,
                                                   gates, G4_, nullptr, 0, G4_, KA_);
        lstm_pointwise<<<(B_ * HID_) / 256, 256, 0, stream>>>(
            gates, ac, ah, xa + 768, KA_, xd, KD_);

        attention_kernel<<<B_, 256, 0, stream>>>(
            ah, wq, vvec, loc_conv, loc_dense, pm, memory, memlen,
            awb, awcb, aligns, t, xa, xd, xp);

        // decoder LSTM gates: [ah|ctx|dh] @ [wih|whh]^T
        wmma_gemm32<<<G4_ / 128, 256, 0, stream>>>(xd, KD_, pkD, bias_d,
                                                   gates, G4_, nullptr, 0, G4_, KD_);
        lstm_pointwise<<<(B_ * HID_) / 256, 256, 0, stream>>>(
            gates, dc, nullptr, xd + 1536, KD_, xp, KP_);

        // proj -> hid (f16), then final -> logits slice of d_out
        wmma_gemm32<<<ENC_ / 128, 256, 0, stream>>>(xp, KP_, pkP, proj_b,
                                                    nullptr, 0, hid, 512, ENC_, KP_);
        wmma_gemm32<<<512 / 128, 256, 0, stream>>>(hid, KF_, pkF, final_b,
                                                   logits + (size_t)t * 512, (long)T_OUT_ * 512,
                                                   nullptr, 0, 512, KF_);
    }
}
